// attentionLayer_32736240730685
// MI455X (gfx1250) — compile-verified, matrix operand order checked by reading, not run on hardware
//
#include <hip/hip_runtime.h>
#include <math.h>

typedef __attribute__((ext_vector_type(16))) _Float16 v16h;
typedef __attribute__((ext_vector_type(8)))  float    v8f;
typedef __attribute__((ext_vector_type(4)))  unsigned int u32x4;
typedef __attribute__((ext_vector_type(8)))  int          i32x8;
typedef __attribute__((ext_vector_type(4)))  int          i32x4;

#define Hc    192
#define Wc    192
#define Cc    64
#define HW    (Hc * Wc)
#define TIL   16
#define PADW  18          // halo tile edge
#define NPIX  256         // 16*16 inner pixels
#define PPIX  324         // 18*18 padded pixels
#define WSTR  72          // halfs per weight row (64 data + 8 pad)
#define KVSTR 66          // floats per pixel in k/v lds ((2p+c)%64 -> conflict-free taps)
#define QSTR  66          // halfs per pixel in q lds
#define KVJOBS 168        // 2 mats * 21 pixel-tiles * 4 o-tiles
#define QJOBS  64         // 16 pixel-tiles * 4 o-tiles

#if defined(__gfx1250__) && __has_builtin(__builtin_amdgcn_tensor_load_to_lds)
#define HAVE_TDM 1
#else
#define HAVE_TDM 0
#endif

__launch_bounds__(256, 1)
__global__ void fused_attn_32736240730685(const float* __restrict__ xg,
                                          const float* __restrict__ Wq, const float* __restrict__ bq,
                                          const float* __restrict__ Wk, const float* __restrict__ bk,
                                          const float* __restrict__ Wv, const float* __restrict__ bv,
                                          float* __restrict__ outg)
{
    // 316,224 B total — fits CDNA5's 320 KB per-workgroup LDS.
    __shared__ __align__(16) float    x_lds[Cc * PPIX];        //  82,944 B  x halo, f32, channel-major [c][p] (TDM layout)
    __shared__ __align__(16) _Float16 w_lds[3 * Cc * WSTR];    //  27,648 B  q/k/v weights, f16
    __shared__ __align__(16) float    kv_lds[2 * PPIX * KVSTR];// 171,072 B  k then v, f32
    __shared__ __align__(16) _Float16 q_lds[NPIX * QSTR];      //  33,792 B  q, f16
    __shared__ float bias_lds[3 * Cc];                         //     768 B

    const int tid  = threadIdx.x;
    const int lane = tid & 31;
    const int wv   = tid >> 5;

    const int bid = blockIdx.x;          // 4 * 12 * 12 tiles
    const int b   = bid / 144;
    const int t   = bid % 144;
    const int h0  = (t / 12) * TIL;
    const int w0  = (t % 12) * TIL;

    const bool interior = (h0 != 0) && (w0 != 0) && (h0 != Hc - TIL) && (w0 != Wc - TIL);

    // ---- phase 1a: x halo tile -> LDS ---------------------------------------
#if HAVE_TDM
    if (interior) {
        if (wv == 0) {   // wave-uniform branch: TDM ignores EXEC, must issue from ONE wave
            unsigned long long ga = (unsigned long long)(size_t)
                (xg + (size_t)b * Cc * HW + (size_t)(h0 - 1) * Wc + (w0 - 1));
            unsigned lds_off = (unsigned)(unsigned long long)
                (__attribute__((address_space(3))) char*)x_lds;
            // D# group0: count=1, lds_addr, 57b global addr, type=2 ("image")
            u32x4 g0;
            g0[0] = 1u;
            g0[1] = lds_off;
            g0[2] = (unsigned)ga;
            g0[3] = (unsigned)((ga >> 32) & 0x01ffffffu) | (2u << 30);
            // D# group1: data_size=4B; tensor dims huge (interior => never OOB);
            // tile 18(x) * 18(y) * 64(z); dim0_stride=192, dim1_stride=192*192
            i32x8 g1;
            g1[0] = (2 << 16);                               // data_size=2 (4 bytes), no pad/iter/mask
            g1[1] = 0;                                       // tensor_dim0[15:0]<<16 (dim0=1<<30)
            g1[2] = (int)((1u << 30) >> 16);                 // dim0[31:16]; dim1[15:0]=0
            g1[3] = (int)(((1u << 30) >> 16) | (18u << 16)); // dim1[31:16] | tile_dim0=18
            g1[4] = (int)(18u | (64u << 16));                // tile_dim1=18 | tile_dim2=64
            g1[5] = (int)Wc;                                 // tensor_dim0_stride = 192
            g1[6] = (int)((unsigned)HW << 16);               // dim1_stride[15:0]=0x9000 in [31:16]
            g1[7] = (int)((unsigned)HW >> 16);               // dim1_stride[47:16] = 0
            // D# group2: tensor_dim2 huge, rest unused (3D tile)
            i32x4 g2; g2[0] = (int)(1u << 30); g2[1] = 0; g2[2] = 0; g2[3] = 0;
            i32x4 g3 = {0, 0, 0, 0};
            i32x8 g4 = {0, 0, 0, 0, 0, 0, 0, 0};             // extended group (clang-23 6-arg form)
            __builtin_amdgcn_tensor_load_to_lds(g0, g1, g2, g3, g4, 0);
        }
    } else
#endif
    {
        // manual reflected copy (border tiles, or no-TDM toolchain), channel-major
        for (int idx = tid; idx < Cc * PPIX; idx += 256) {
            int c = idx / PPIX, p = idx - c * PPIX;
            int ph = p / PADW, pw = p - ph * PADW;
            int gh = h0 - 1 + ph; gh = (gh < 0) ? -gh : ((gh >= Hc) ? 2 * Hc - 2 - gh : gh);
            int gw = w0 - 1 + pw; gw = (gw < 0) ? -gw : ((gw >= Wc) ? 2 * Wc - 2 - gw : gw);
            x_lds[idx] = xg[(size_t)b * Cc * HW + (size_t)c * HW + (size_t)gh * Wc + gw];
        }
    }

    // ---- phase 1b: weights (f16) + bias -> LDS ------------------------------
    for (int idx = tid; idx < 3 * Cc * Cc; idx += 256) {
        int m = idx >> 12;                       // 0:q 1:k 2:v
        int r = idx & 4095;
        int o = r >> 6, c = r & 63;
        const float* ws = (m == 0) ? Wq : (m == 1 ? Wk : Wv);
        w_lds[(m * Cc + o) * WSTR + c] = (_Float16)ws[r];
    }
    if (tid < 3 * Cc) {
        int m = tid >> 6;
        const float* bs = (m == 0) ? bq : (m == 1 ? bk : bv);
        bias_lds[tid] = bs[tid & 63];
    }

#if HAVE_TDM
    if (interior && wv == 0) {
#if __has_builtin(__builtin_amdgcn_s_wait_tensorcnt)
        __builtin_amdgcn_s_wait_tensorcnt(0);
#else
        asm volatile("s_wait_tensorcnt 0x0" ::: "memory");
#endif
    }
#endif
    __syncthreads();

    // ---- phase 2: WMMA GEMMs for q / k / v ----------------------------------
    // job j in [0,168): k/v over padded pixels; j in [168,232): q over inner pixels.
    const int koff = (lane < 16) ? 0 : 8;     // K-offset per documented 16-bit A/B striping
    const int nloc = lane & 15;               // N (pixel) index within tile
    for (int j = wv; j < KVJOBS + QJOBS; j += 8) {
        int ot, mbias, srcpix, dsti, is_q, valid = 1;
        if (j < KVJOBS) {
            int m12 = j / 84;                 // 0:k 1:v
            int rem = j % 84;
            int pt  = rem >> 2; ot = rem & 3;
            int p   = pt * 16 + nloc;
            valid   = (p < PPIX);
            if (!valid) p = PPIX - 1;
            srcpix  = p;
            is_q    = 0;
            dsti    = (m12 * PPIX + p) * KVSTR;
            mbias   = 1 + m12;
        } else {
            int rem = j - KVJOBS;
            int pt  = rem >> 2; ot = rem & 3;
            int ip  = pt * 16 + nloc;                         // inner pixel 0..255
            srcpix  = ((ip >> 4) + 1) * PADW + ((ip & 15) + 1);
            is_q    = 1;
            dsti    = ip * QSTR;
            mbias   = 0;
        }

        // A fragments: f16 weights, direct (M = out-channel, K = in-channel)
        const _Float16* wb = w_lds + (mbias * Cc + ot * 16 + nloc) * WSTR + koff;
        // B fragments: f32 x, channel-major -> convert to f16 (K = channel, N = pixel)
        const float*    xb = x_lds + (size_t)koff * PPIX + srcpix;
        v16h a0, a1, b0, b1;
        #pragma unroll
        for (int e = 0; e < 8; ++e) {
            a0[e]     = wb[e];      a0[e + 8] = wb[e + 16];
            a1[e]     = wb[32 + e]; a1[e + 8] = wb[32 + e + 16];
            b0[e]     = (_Float16)xb[(size_t)e * PPIX];
            b0[e + 8] = (_Float16)xb[(size_t)(e + 16) * PPIX];
            b1[e]     = (_Float16)xb[(size_t)(e + 32) * PPIX];
            b1[e + 8] = (_Float16)xb[(size_t)(e + 48) * PPIX];
        }
        v8f acc = {};
        acc = __builtin_amdgcn_wmma_f32_16x16x32_f16(false, a0, false, b0,
                                                     (short)0, acc, false, false);
        acc = __builtin_amdgcn_wmma_f32_16x16x32_f16(false, a1, false, b1,
                                                     (short)0, acc, false, false);
        if (valid) {
            int cbase = ot * 16 + ((lane >= 16) ? 8 : 0);     // D: M = r + (hi-half? 8)
            if (is_q) {
                _Float16* d = q_lds + dsti;
                #pragma unroll
                for (int r = 0; r < 8; ++r)
                    d[cbase + r] = (_Float16)(acc[r] + bias_lds[cbase + r]);      // mbias==0
            } else {
                float* d = kv_lds + dsti;
                #pragma unroll
                for (int r = 0; r < 8; ++r)
                    d[cbase + r] = acc[r] + bias_lds[mbias * Cc + cbase + r];
            }
        }
    }
    __syncthreads();

    // ---- phase 3: per-pixel per-channel 9-tap softmax (k/v f32: no converts) -
    const int i  = tid >> 4;
    const int j2 = tid & 15;
    const float* kb = kv_lds;
    const float* vb = kv_lds + PPIX * KVSTR;
    float* op = outg + (size_t)b * Cc * HW + (size_t)(h0 + i) * Wc + (w0 + j2);
    #pragma unroll 2
    for (int c = 0; c < Cc; ++c) {
        float qv = (float)q_lds[tid * QSTR + c];
        float s[9];
        float mx = -1e30f;
        #pragma unroll
        for (int dy = 0; dy < 3; ++dy)
            #pragma unroll
            for (int dx = 0; dx < 3; ++dx) {
                int p = (i + dy) * PADW + (j2 + dx);
                float sv = qv * kb[p * KVSTR + c];
                s[dy * 3 + dx] = sv;
                mx = fmaxf(mx, sv);
            }
        float sum = 0.f, accv = 0.f;
        #pragma unroll
        for (int dy = 0; dy < 3; ++dy)
            #pragma unroll
            for (int dx = 0; dx < 3; ++dx) {
                int p = (i + dy) * PADW + (j2 + dx);
                float e = __expf(s[dy * 3 + dx] - mx);
                sum  += e;
                accv += e * vb[p * KVSTR + c];
            }
        op[(size_t)c * HW] = accv / sum;
    }
}

extern "C" void kernel_launch(void* const* d_in, const int* in_sizes, int n_in,
                              void* d_out, int out_size, void* d_ws, size_t ws_size,
                              hipStream_t stream) {
    const float* x  = (const float*)d_in[0];
    const float* Wq = (const float*)d_in[1];
    const float* bq = (const float*)d_in[2];
    const float* Wk = (const float*)d_in[3];
    const float* bk = (const float*)d_in[4];
    const float* Wv = (const float*)d_in[5];
    const float* bv = (const float*)d_in[6];
    float* out = (float*)d_out;

    dim3 grid(4 * 12 * 12);   // one WG per 16x16 tile per batch image
    dim3 block(256);          // 8 wave32
    fused_attn_32736240730685<<<grid, block, 0, stream>>>(x, Wq, bq, Wk, bk, Wv, bv, out);
}
